// PartialEncoderWeightedSumEDDIMultiWeight_6846177870203
// MI455X (gfx1250) — compile-verified, matrix operand order checked by reading, not run on hardware
//
#include <hip/hip_runtime.h>
#include <hip/hip_bf16.h>

typedef __attribute__((ext_vector_type(16))) __bf16 v16bf;
typedef __attribute__((ext_vector_type(8)))  float  v8f;
typedef __attribute__((ext_vector_type(4))) unsigned int ui32x4;
typedef __attribute__((ext_vector_type(8))) int i32x8;
typedef __attribute__((ext_vector_type(4))) int i32x4;

#define B_  128
#define J_  8192
#define D_  32
#define W_  4
#define H_  128
#define EH_ 256
#define TJ  128      // j-tile per iteration
#define NT  (J_ / TJ)
#define H1S 136      // bf16 row stride for h1 / w2T tiles (banks + 16B align)
#define HOS 33       // f32 row stride for h_out tile
#define FEP 129      // f32 row stride for TDM-landed fejh tile (bank-conflict-free)

// ---------------------------------------------------------------------------
// Prologue: fejhb[j][h] = (fe @ h_w1[1:])[j][h] + h_b1[h]   (shared across B)
// ---------------------------------------------------------------------------
__global__ void eddi_fejh_kernel(const float* __restrict__ fe,
                                 const float* __restrict__ h_w1,
                                 const float* __restrict__ h_b1,
                                 float* __restrict__ out) {
    int idx = blockIdx.x * blockDim.x + threadIdx.x;   // J_*H_ threads
    int j = idx >> 7, h = idx & 127;
    float acc = h_b1[h];
#pragma unroll
    for (int d = 0; d < D_; ++d)
        acc += fe[j * D_ + d] * h_w1[(1 + d) * H_ + h];
    out[idx] = acc;
}

// Load a 16x32 bf16 fragment (A or B-transposed) from LDS, row-major,
// per the CDNA5 16-bit A/B VGPR layout (ISA 7.12.2).
__device__ __forceinline__ v16bf load_frag16(const __bf16* base, int row0,
                                             int kbase, int lane) {
    int r  = row0 + (lane & 15);
    int k0 = kbase + ((lane >> 4) << 3);
    const __bf16* p = base + r * H1S + k0;
    union { uint4 q[2]; v16bf v; } u;
    u.q[0] = *(const uint4*)(p);        // K k0..k0+7
    u.q[1] = *(const uint4*)(p + 16);   // K k0+16..k0+23
    return u.v;
}

// Issue one TDM load: 128x128 f32 tile, row stride 128 f32 in memory,
// landed in LDS with +1 dword padding every 128 dwords (row stride FEP=129).
// D# per CDNA5 ISA §8: group0 {count=1, lds_addr, global_addr, type=2},
// group1 {data_size=4B, pad_enable, pad_interval=128dw, pad_amount=1dw,
//         tensor_dim=128x128, tile_dim=128x128, dim0_stride=128}.
__device__ __forceinline__ void tdm_load_tile(const void* gsrc, unsigned lds_off) {
    unsigned long long ga = (unsigned long long)(uintptr_t)gsrc;
    ui32x4 g0;
    g0[0] = 1u;                                    // count=1, user mode, no gather
    g0[1] = lds_off;                               // lds_addr (bytes)
    g0[2] = (unsigned)ga;                          // global_addr[31:0]
    g0[3] = (unsigned)(ga >> 32) | (2u << 30);     // global_addr[56:32] | type=2
    i32x8 g1;
    g1[0] = (2 << 16) | (1 << 20) | (6 << 22);     // data_size=4B | pad_en | ivl=128dw | amt=1dw
    g1[1] = (TJ << 16);                            // tensor_dim0[15:0] << 16
    g1[2] = (TJ << 16);                            // dim0 hi=0 | tensor_dim1[15:0] << 16
    g1[3] = (TJ << 16);                            // dim1 hi=0 | tile_dim0 << 16
    g1[4] = TJ;                                    // tile_dim1 (tile_dim2=0)
    g1[5] = H_;                                    // tensor_dim0_stride = 128 elems
    g1[6] = 0;
    g1[7] = 0;
    i32x4 z4 = {0, 0, 0, 0};
    i32x8 z8 = {0, 0, 0, 0, 0, 0, 0, 0};
    __builtin_amdgcn_tensor_load_to_lds(g0, g1, z4, z4, z8, 0);
}

// ---------------------------------------------------------------------------
// Main kernel: one block per batch b, 256 threads = 8 wave32
// ---------------------------------------------------------------------------
__launch_bounds__(256)
__global__ void eddi_cell_kernel(
    const float* __restrict__ x, const int* __restrict__ mask,
    const float* __restrict__ fejhb,
    const float* __restrict__ h_w1,
    const float* __restrict__ h_ln1_w, const float* __restrict__ h_ln1_b,
    const float* __restrict__ h_w2,    const float* __restrict__ h_b2,
    const float* __restrict__ h_ln2_w, const float* __restrict__ h_ln2_b,
    const float* __restrict__ g_w1, const float* __restrict__ g_b1,
    const float* __restrict__ g_w2, const float* __restrict__ g_b2,
    const float* __restrict__ c_w,  const float* __restrict__ c_b,
    const float* __restrict__ c_ln_w, const float* __restrict__ c_ln_b,
    const float* __restrict__ e_w1, const float* __restrict__ e_b1,
    const float* __restrict__ e_ln1_w, const float* __restrict__ e_ln1_b,
    const float* __restrict__ e_w2, const float* __restrict__ e_b2,
    const float* __restrict__ e_ln2_w, const float* __restrict__ e_ln2_b,
    float* __restrict__ out)
{
    extern __shared__ __align__(16) char smem[];
    float*  s_fe0 = (float*)smem;                // TDM buf 0: TJ x FEP f32
    float*  s_fe1 = s_fe0 + TJ * FEP;            // TDM buf 1
    __bf16* s_h1  = (__bf16*)(s_fe1 + TJ * FEP); // TJ  x H1S  (LN1+ReLU, bf16)
    __bf16* s_w2t = s_h1 + TJ * H1S;             // 32  x H1S  (h_w2 transposed)
    float*  s_hout = (float*)(s_w2t + 32 * H1S); // TJ  x HOS
    float*  s_p   = s_hout + TJ * HOS;           // TJ x 4 (exp weights); tail reuse
    float*  s_w0  = s_p + TJ * 4;                // 128
    float*  s_l1w = s_w0 + H_;                   // 128
    float*  s_l1b = s_l1w + H_;                  // 128
    float*  s_b2  = s_l1b + H_;                  // 32
    float*  s_l2w = s_b2 + D_;                   // 32
    float*  s_l2b = s_l2w + D_;                  // 32
    float*  s_g1  = s_l2b + D_;                  // 512
    float*  s_gb1 = s_g1 + 512;                  // 16
    float*  s_g2  = s_gb1 + 16;                  // 64
    float*  s_gb2 = s_g2 + 64;                   // 4
    float*  s_acc = s_gb2 + 4;                   // 128 : A[w][d]
    float*  s_sw  = s_acc + 128;                 // 4   : sum_j exp
    float*  s_t0  = s_sw + 4;                    // 256 : tail scratch
    float*  s_t1  = s_t0 + 256;                  // 64  : tail scratch
    float*  s_red = s_t1 + 64;                   // 8   : reduction scalars

    const int tid  = threadIdx.x;
    const int wv   = tid >> 5;      // wave id 0..7 -> M tile
    const int lane = tid & 31;
    const int b    = blockIdx.x;

    const unsigned fe_off0 = (unsigned)(uintptr_t)(void*)s_fe0;
    const unsigned fe_off1 = (unsigned)(uintptr_t)(void*)s_fe1;

    // -------- preload weights into LDS --------
    for (int i = tid; i < H_; i += 256) {
        s_w0[i]  = h_w1[i];          // row 0 of h_w1 = scalar-x column weights
        s_l1w[i] = h_ln1_w[i];
        s_l1b[i] = h_ln1_b[i];
    }
    if (tid < D_) { s_b2[tid] = h_b2[tid]; s_l2w[tid] = h_ln2_w[tid]; s_l2b[tid] = h_ln2_b[tid]; }
    for (int i = tid; i < 512; i += 256) s_g1[i] = g_w1[i];
    if (tid < 16) s_gb1[tid] = g_b1[tid];
    if (tid < 64) s_g2[tid]  = g_w2[tid];
    if (tid < 4)  s_gb2[tid] = g_b2[tid];
    for (int i = tid; i < H_ * D_; i += 256) {       // w2T[n][k] = h_w2[k][n]
        int k = i >> 5, n = i & 31;
        s_w2t[n * H1S + k] = (__bf16)h_w2[i];
    }
    if (tid < 128) s_acc[tid] = 0.f;
    if (tid < 4)   s_sw[tid]  = 0.f;

    // -------- prime the TDM double-buffer pipeline (wave 0 only) --------
    if (wv == 0) {
        tdm_load_tile(fejhb + (size_t)0 * TJ * H_, fe_off0);
        tdm_load_tile(fejhb + (size_t)1 * TJ * H_, fe_off1);
    }
    __syncthreads();

    // -------- hoist B fragments (h_w2ᵀ is loop-invariant) --------
    v16bf bfr[2][4];
#pragma unroll
    for (int ni = 0; ni < 2; ++ni)
#pragma unroll
        for (int kk = 0; kk < 4; ++kk)
            bfr[ni][kk] = load_frag16(s_w2t, ni * 16, kk * 32, lane);

    const float* xb = x    + (size_t)b * J_;
    const int*   mb = mask + (size_t)b * J_;

    for (int t = 0; t < NT; ++t) {
        const int j0 = t * TJ;
        float* s_fe = (t & 1) ? s_fe1 : s_fe0;

        // ---- wait for tile t's TDM (oldest of the two outstanding) ----
        if (wv == 0) {
            if (t < NT - 1) __builtin_amdgcn_s_wait_tensorcnt(1);
            else            __builtin_amdgcn_s_wait_tensorcnt(0);
        }
        __syncthreads();

        // ---- phase 1+2 fused: pre = x*w0 + fejh ; LN over H=128 ; ReLU -> bf16 ----
        if (tid < TJ) {
            const float* fr = s_fe + tid * FEP;
            float xv = xb[j0 + tid];
            float sum = 0.f, sq = 0.f;
#pragma unroll 4
            for (int h = 0; h < H_; ++h) {
                float v = fmaf(xv, s_w0[h], fr[h]);
                sum += v; sq += v * v;
            }
            float mu  = sum * (1.f / H_);
            float var = sq * (1.f / H_) - mu * mu;
            float inv = rsqrtf(var + 1e-5f);
            __bf16* wrow = s_h1 + tid * H1S;
#pragma unroll 4
            for (int h = 0; h < H_; ++h) {
                float v = fmaf(xv, s_w0[h], fr[h]);
                v = (v - mu) * inv * s_l1w[h] + s_l1b[h];
                wrow[h] = (__bf16)fmaxf(v, 0.f);
            }
        }
        __syncthreads();

        // ---- kick TDM for tile t+2 into the buffer we just consumed ----
        if (wv == 0 && t + 2 < NT)
            tdm_load_tile(fejhb + (size_t)(t + 2) * TJ * H_,
                          (t & 1) ? fe_off1 : fe_off0);

        // ---- phase 3: h_out_pre = h1 @ h_w2 via WMMA (128x128x32) ----
        {
            v16bf afr[4];
#pragma unroll
            for (int kk = 0; kk < 4; ++kk)
                afr[kk] = load_frag16(s_h1, wv * 16, kk * 32, lane);
#pragma unroll
            for (int ni = 0; ni < 2; ++ni) {
                v8f acc = {};
#pragma unroll
                for (int kk = 0; kk < 4; ++kk)
                    acc = __builtin_amdgcn_wmma_f32_16x16x32_bf16(
                        false, afr[kk], false, bfr[ni][kk],
                        (short)0, acc, false, false);
                int nloc = lane & 15, mb2 = (lane >> 4) * 8;
#pragma unroll
                for (int v = 0; v < 8; ++v)
                    s_hout[(wv * 16 + mb2 + v) * HOS + ni * 16 + nloc] = acc[v];
            }
        }
        __syncthreads();

        // ---- phase 4: +bias, LN over D=32, ReLU, gate MLP, exp weights ----
        if (tid < TJ) {
            float row[32];
            float sum = 0.f, sq = 0.f;
#pragma unroll
            for (int d = 0; d < D_; ++d) {
                float v = s_hout[tid * HOS + d] + s_b2[d];
                row[d] = v; sum += v; sq += v * v;
            }
            float mu  = sum * (1.f / D_);
            float var = sq * (1.f / D_) - mu * mu;
            float inv = rsqrtf(var + 1e-5f);
#pragma unroll
            for (int d = 0; d < D_; ++d) {
                float v = (row[d] - mu) * inv * s_l2w[d] + s_l2b[d];
                row[d] = fmaxf(v, 0.f);
                s_hout[tid * HOS + d] = row[d];
            }
            float lg[4] = { s_gb2[0], s_gb2[1], s_gb2[2], s_gb2[3] };
#pragma unroll
            for (int k = 0; k < 16; ++k) {
                float tacc = s_gb1[k];
#pragma unroll
                for (int d = 0; d < D_; ++d) tacc += row[d] * s_g1[d * 16 + k];
                tacc = fmaxf(tacc, 0.f);
#pragma unroll
                for (int w = 0; w < W_; ++w) lg[w] += tacc * s_g2[k * 4 + w];
            }
            int m = mb[j0 + tid];
#pragma unroll
            for (int w = 0; w < W_; ++w) {
                // clip to [-10,10] -> exp() safe without max-subtraction
                float l = fminf(fmaxf(lg[w], -10.f), 10.f);
                s_p[tid * 4 + w] = (m > 0) ? __expf(l) : 0.f;
            }
        }
        __syncthreads();

        // ---- phase 5: streaming accumulate  A[w][d] += p·h_out,  s[w] += p ----
        if (tid < 128) {
            int w = tid >> 5, d = tid & 31;
            float a = 0.f;
#pragma unroll 4
            for (int jj = 0; jj < TJ; ++jj)
                a += s_p[jj * 4 + w] * s_hout[jj * HOS + d];
            s_acc[tid] += a;
        } else if (tid < 132) {
            int w = tid - 128;
            float sm = 0.f;
#pragma unroll 4
            for (int jj = 0; jj < TJ; ++jj) sm += s_p[jj * 4 + w];
            s_sw[w] += sm;
        }
        // next iteration's top barrier separates these LDS reads/writes
    }
    __syncthreads();

    // ======================= per-batch tail MLP =======================
    if (tid < 128) {                                  // head_flat[t], t = w*32+d
        float sw = s_sw[tid >> 5];
        s_p[tid] = (sw > 0.f) ? s_acc[tid] / sw : 0.f;
    }
    __syncthreads();
    if (tid < 32) {                                   // comb_pre = head @ c_w + c_b
        float cp = c_b[tid];
        for (int t = 0; t < 128; ++t) cp += s_p[t] * c_w[t * 32 + tid];
        s_p[128 + tid] = cp;
    }
    __syncthreads();
    if (tid == 0) {
        float sum = 0.f, sq = 0.f;
        for (int i = 0; i < 32; ++i) { float v = s_p[128 + i]; sum += v; sq += v * v; }
        float mu = sum / 32.f;
        s_red[0] = mu; s_red[1] = rsqrtf(sq / 32.f - mu * mu + 1e-5f);
    }
    __syncthreads();
    if (tid < 32) {
        float v = (s_p[128 + tid] - s_red[0]) * s_red[1] * c_ln_w[tid] + c_ln_b[tid];
        v = fmaxf(v, 0.f);
        if (!(s_sw[0] > 0.f)) v = 0.f;                // empty cell -> zeros
        s_p[160 + tid] = v;                           // comb
    }
    __syncthreads();
    {                                                 // e1: 32 -> 256
        float p2 = e_b1[tid];
#pragma unroll
        for (int d = 0; d < 32; ++d) p2 += s_p[160 + d] * e_w1[d * 256 + tid];
        s_t0[tid] = p2;
    }
    __syncthreads();
    if (tid == 0) {
        float sum = 0.f, sq = 0.f;
        for (int i = 0; i < 256; ++i) { float v = s_t0[i]; sum += v; sq += v * v; }
        float mu = sum / 256.f;
        s_red[2] = mu; s_red[3] = rsqrtf(sq / 256.f - mu * mu + 1e-5f);
    }
    __syncthreads();
    {
        float v = (s_t0[tid] - s_red[2]) * s_red[3] * e_ln1_w[tid] + e_ln1_b[tid];
        s_t0[tid] = fmaxf(v, 0.f);
    }
    __syncthreads();
    if (tid < 64) {                                   // e2: 256 -> 64
        float op = e_b2[tid];
        for (int e = 0; e < 256; ++e) op += s_t0[e] * e_w2[e * 64 + tid];
        s_t1[tid] = op;
    }
    __syncthreads();
    if (tid == 0) {
        float sum = 0.f, sq = 0.f;
        for (int i = 0; i < 64; ++i) { float v = s_t1[i]; sum += v; sq += v * v; }
        float mu = sum / 64.f;
        s_red[4] = mu; s_red[5] = rsqrtf(sq / 64.f - mu * mu + 1e-5f);
    }
    __syncthreads();
    if (tid < 64) {
        float v = (s_t1[tid] - s_red[4]) * s_red[5] * e_ln2_w[tid] + e_ln2_b[tid];
        v = fmaxf(v, 0.f);
        if (tid < 32) out[(size_t)b * 32 + tid] = v;                 // mu
        else          out[4096 + (size_t)b * 32 + (tid - 32)] = v;   // logvar
    }
}

extern "C" void kernel_launch(void* const* d_in, const int* in_sizes, int n_in,
                              void* d_out, int out_size, void* d_ws, size_t ws_size,
                              hipStream_t stream) {
    const float* x       = (const float*)d_in[0];
    const int*   mask    = (const int*)  d_in[1];
    const float* fe      = (const float*)d_in[2];
    const float* h_w1    = (const float*)d_in[3];
    const float* h_b1    = (const float*)d_in[4];
    const float* h_ln1_w = (const float*)d_in[5];
    const float* h_ln1_b = (const float*)d_in[6];
    const float* h_w2    = (const float*)d_in[7];
    const float* h_b2    = (const float*)d_in[8];
    const float* h_ln2_w = (const float*)d_in[9];
    const float* h_ln2_b = (const float*)d_in[10];
    const float* g_w1    = (const float*)d_in[11];
    const float* g_b1    = (const float*)d_in[12];
    const float* g_w2    = (const float*)d_in[13];
    const float* g_b2    = (const float*)d_in[14];
    const float* c_w     = (const float*)d_in[15];
    const float* c_b     = (const float*)d_in[16];
    const float* c_ln_w  = (const float*)d_in[17];
    const float* c_ln_b  = (const float*)d_in[18];
    const float* e_w1    = (const float*)d_in[19];
    const float* e_b1    = (const float*)d_in[20];
    const float* e_ln1_w = (const float*)d_in[21];
    const float* e_ln1_b = (const float*)d_in[22];
    const float* e_w2    = (const float*)d_in[23];
    const float* e_b2    = (const float*)d_in[24];
    const float* e_ln2_w = (const float*)d_in[25];
    const float* e_ln2_b = (const float*)d_in[26];

    float* fejhb = (float*)d_ws;   // J_*H_ floats = 4 MB (L2-resident)

    eddi_fejh_kernel<<<(J_ * H_) / 256, 256, 0, stream>>>(fe, h_w1, h_b1, fejhb);

    constexpr size_t SMEM_BYTES =
        (size_t)(2 * TJ * FEP) * 4 +                       // TDM double buffer
        (size_t)(TJ * H1S + 32 * H1S) * 2 +                // bf16 tiles
        (size_t)(TJ * HOS + TJ * 4 + 3 * H_ + 3 * D_ + 512 + 16 + 64 + 4 +
                 128 + 4 + 256 + 64 + 8) * 4;

    eddi_cell_kernel<<<B_, 256, SMEM_BYTES, stream>>>(
        x, mask, fejhb, h_w1, h_ln1_w, h_ln1_b, h_w2, h_b2, h_ln2_w, h_ln2_b,
        g_w1, g_b1, g_w2, g_b2, c_w, c_b, c_ln_w, c_ln_b,
        e_w1, e_b1, e_ln1_w, e_ln1_b, e_w2, e_b2, e_ln2_w, e_ln2_b,
        (float*)d_out);
}
